// TPS_STN_38886633898226
// MI455X (gfx1250) — compile-verified, matrix operand order checked by reading, not run on hardware
//
#include <hip/hip_runtime.h>
#include <hip/hip_bf16.h>

typedef __attribute__((ext_vector_type(16))) _Float16 v16h;
typedef __attribute__((ext_vector_type(8)))  _Float16 v8h;
typedef __attribute__((ext_vector_type(8)))  float    v8f;

// ---------------------------------------------------------------------------
// K-slot mapping for 16-bit WMMA fragments (ISA 7.12.2, 16-bit 16x32):
//   half slot h (0..15): VGPRs 0..3 cover K = kb..kb+7, VGPRs 4..7 cover
//   K = 16+kb..16+kb+7, kb = 8*(lane>>4).  A lane's fragment is two
//   contiguous 8-half (16B) K-spans: [kb, kb+8) and [16+kb, 16+kb+8).
// ---------------------------------------------------------------------------
__device__ __forceinline__ int wmma_k_slot(int hi, int h) {
    int v = h >> 1, j = h & 1;
    int base = (v < 4) ? (2 * v + j) : (16 + 2 * (v - 4) + j);
    return base + hi * 8;
}

__device__ __forceinline__ v16h cat16(v8h a, v8h b) {
    return __builtin_shufflevector(a, b, 0, 1, 2, 3, 4, 5, 6, 7,
                                   8, 9, 10, 11, 12, 13, 14, 15);
}

__device__ __forceinline__ v8f wmma_f16(v16h a, v16h b, v8f c) {
    return __builtin_amdgcn_wmma_f32_16x16x32_f16(false, a, false, b,
                                                  (short)0, c, false, false);
}

// ---------------------------------------------------------------------------
// Zero-fill f16 buffer, 8 halfs (16B) per thread
// ---------------------------------------------------------------------------
__global__ void zero_v8h_kernel(v8h* __restrict__ p, long n8) {
    long i = (long)blockIdx.x * blockDim.x + threadIdx.x;
    if (i < n8) p[i] = (v8h)(_Float16)0.0f;
}

// ---------------------------------------------------------------------------
// fp32 NHWC -> fp16 NHWC with 1-pixel spatial halo (halo pre-zeroed)
// ---------------------------------------------------------------------------
__global__ void pad_cvt_kernel(const float* __restrict__ in, _Float16* __restrict__ out,
                               int B, int H, int W, int C) {
    int idx = blockIdx.x * blockDim.x + threadIdx.x;
    if (idx >= B * H * W * C) return;
    int c = idx % C, t = idx / C;
    int x = t % W;  t /= W;
    int y = t % H;  t /= H;
    int b = t;
    out[(((long)b * (H + 2) + (y + 1)) * (W + 2) + (x + 1)) * C + c] = (_Float16)in[idx];
}

// ---------------------------------------------------------------------------
// Swizzle conv weights (HWIO fp32, flat [K, Cout], K = 9*Cin) into WMMA
// B-fragment order: [nt16][ktile][lane][16 halfs], K zero-padded to 32*nkt.
// Conv kernels then fetch B fragments as two contiguous b128 loads.
// ---------------------------------------------------------------------------
__global__ void wswz_kernel(const float* __restrict__ w, _Float16* __restrict__ out,
                            int K, int Cout, int nkt) {
    int idx = blockIdx.x * blockDim.x + threadIdx.x;
    int total = (Cout >> 4) * nkt * 32 * 16;
    if (idx >= total) return;
    int h = idx & 15;
    int lane = (idx >> 4) & 31;
    int rest = idx >> 9;
    int ktile = rest % nkt, nt = rest / nkt;
    int lo = lane & 15, hi = lane >> 4;
    int kk = ktile * 32 + wmma_k_slot(hi, h);
    int n = nt * 16 + lo;
    out[idx] = (kk < K) ? (_Float16)w[kk * Cout + n] : (_Float16)0.0f;
}

// ---------------------------------------------------------------------------
// Layer 1: conv3x3 (Cin=3, Cout=64) + BN + ReLU + fused 2x2 maxpool.
// K=27 (padded to 32).  A halves are unconditional clamped gathers from the
// haloed input; K-padding correctness comes from the zero-padded weights
// (a_val * 0 = 0), so there are no branches and loads batch under one wait.
// ---------------------------------------------------------------------------
__global__ __launch_bounds__(256)
void conv1_kernel(const _Float16* __restrict__ act,    // [64][66][258][3]
                  const _Float16* __restrict__ wswz,   // [4][1][32][16]
                  const float* __restrict__ bn_g, const float* __restrict__ bn_b,
                  const float* __restrict__ bn_m, const float* __restrict__ bn_v,
                  _Float16* __restrict__ out) {        // [64][34][130][64]
    const int lane = threadIdx.x & 31;
    const int lo = lane & 15, hi = lane >> 4;
    const int wave = (blockIdx.x * blockDim.x + threadIdx.x) >> 5;
    // waves = B(64) * Hout(32) * cgroups(16) * ntiles(4)
    int t = wave;
    const int nt  = t & 3;   t >>= 2;
    const int cg  = t & 15;  t >>= 4;
    const int row = t & 31;  t >>= 5;
    const int b   = t;

    const int n  = nt * 16 + lo;
    const int xc = cg * 16;
    const int y0 = 2 * row;

    const _Float16* wp = wswz + ((size_t)nt * 32 + lane) * 16;
    v16h bfrag = cat16(*(const v8h*)wp, *(const v8h*)(wp + 8));

    v16h a0, a1;
    #pragma unroll
    for (int h = 0; h < 16; ++h) {
        int kk = wmma_k_slot(hi, h);
        int kkc = (kk > 26) ? 26 : kk;           // clamp for addressing only
        int ci = kkc % 3, r = kkc / 3;
        int kx = r % 3,   ky = r / 3;
        long base = (((long)b * 66 + (y0 + ky)) * 258 + (xc + lo + kx)) * 3 + ci;
        a0[h] = act[base];
        a1[h] = act[base + 258 * 3];
    }
    v8f acc0 = {}, acc1 = {};
    acc0 = wmma_f16(a0, bfrag, acc0);
    acc1 = wmma_f16(a1, bfrag, acc1);

    const float sc = bn_g[n] * rsqrtf(bn_v[n] + 1e-3f);
    const float bi = bn_b[n] - bn_m[n] * sc;
    #pragma unroll
    for (int p = 0; p < 4; ++p) {
        float v00 = fmaxf(acc0[2 * p]     * sc + bi, 0.f);
        float v01 = fmaxf(acc0[2 * p + 1] * sc + bi, 0.f);
        float v10 = fmaxf(acc1[2 * p]     * sc + bi, 0.f);
        float v11 = fmaxf(acc1[2 * p + 1] * sc + bi, 0.f);
        float mx  = fmaxf(fmaxf(v00, v01), fmaxf(v10, v11));
        int pc = (xc >> 1) + 4 * hi + p;
        out[(((long)b * 34 + (row + 1)) * 130 + (pc + 1)) * 64 + n] = (_Float16)mx;
    }
}

// ---------------------------------------------------------------------------
// Layers 2-3: conv3x3 + BN + ReLU + fused 2x2 maxpool, haloed in/out.
// Wide tile: 2 conv rows x 16 cols x 32 out channels per wave (4 accums).
// Per K-step: 4 A-loads + 4 B-loads feed 4 WMMAs (2 b128 loads / WMMA).
// ---------------------------------------------------------------------------
__global__ __launch_bounds__(256)
void conv_pool_kernel(const _Float16* __restrict__ act,   // [B][Hin+2][Win+2][Cin]
                      const _Float16* __restrict__ wswz,  // [Cout/16][nkt][32][16]
                      const float* __restrict__ bn_g, const float* __restrict__ bn_b,
                      const float* __restrict__ bn_m, const float* __restrict__ bn_v,
                      _Float16* __restrict__ out,         // [B][Hout+2][Wout+2][Cout]
                      int B, int Hin, int Win, int Cin, int Cout) {
    const int Hout = Hin >> 1, Wout = Win >> 1;
    const int Wp = Win + 2;
    const int lane = threadIdx.x & 31;
    const int lo = lane & 15, hi = lane >> 4;
    const int kb = hi * 8;
    const int wave = (blockIdx.x * blockDim.x + threadIdx.x) >> 5;

    const int ntiles  = Cout >> 5;      // 32 channels per wave
    const int cgroups = Win >> 4;
    int t = wave;
    const int nt  = t % ntiles;  t /= ntiles;
    const int cg  = t % cgroups; t /= cgroups;
    const int row = t % Hout;    t /= Hout;
    const int b   = t;
    if (b >= B) return;

    const int n0 = nt * 32 + lo, n1 = n0 + 16;
    const int xc = cg * 16;
    const int y0 = 2 * row;
    const int nkt = (9 * Cin) >> 5;

    const _Float16* wb0 = wswz + ((size_t)(2 * nt) * nkt * 32 + lane) * 16;
    const _Float16* wb1 = wb0 + (size_t)nkt * 512;

    v8f acc00 = {}, acc01 = {}, acc10 = {}, acc11 = {};
    int ktile = 0;
    for (int pos = 0; pos < 9; ++pos) {
        const int ky = pos / 3, kx = pos % 3;
        const _Float16* arow0 =
            act + (((size_t)b * (Hin + 2) + (y0 + ky)) * Wp + (xc + lo + kx)) * Cin;
        const _Float16* arow1 = arow0 + (size_t)Wp * Cin;
        for (int ct = 0; ct < Cin; ct += 32, ++ktile) {
            const _Float16* wp0 = wb0 + (size_t)ktile * 512;
            const _Float16* wp1 = wb1 + (size_t)ktile * 512;
            __builtin_prefetch(wp0 + 512, 0, 1);
            v16h bf0 = cat16(*(const v8h*)wp0, *(const v8h*)(wp0 + 8));
            v16h bf1 = cat16(*(const v8h*)wp1, *(const v8h*)(wp1 + 8));

            v16h a0 = cat16(*(const v8h*)(arow0 + ct + kb),
                            *(const v8h*)(arow0 + ct + 16 + kb));
            v16h a1 = cat16(*(const v8h*)(arow1 + ct + kb),
                            *(const v8h*)(arow1 + ct + 16 + kb));
            acc00 = wmma_f16(a0, bf0, acc00);
            acc01 = wmma_f16(a0, bf1, acc01);
            acc10 = wmma_f16(a1, bf0, acc10);
            acc11 = wmma_f16(a1, bf1, acc11);
        }
    }

    const int Hd = Hout + 2, Wd = Wout + 2;
    #pragma unroll
    for (int g = 0; g < 2; ++g) {
        const int n = g ? n1 : n0;
        const v8f& r0 = g ? acc01 : acc00;
        const v8f& r1 = g ? acc11 : acc10;
        const float sc = bn_g[n] * rsqrtf(bn_v[n] + 1e-3f);
        const float bi = bn_b[n] - bn_m[n] * sc;
        #pragma unroll
        for (int p = 0; p < 4; ++p) {
            float v00 = fmaxf(r0[2 * p]     * sc + bi, 0.f);
            float v01 = fmaxf(r0[2 * p + 1] * sc + bi, 0.f);
            float v10 = fmaxf(r1[2 * p]     * sc + bi, 0.f);
            float v11 = fmaxf(r1[2 * p + 1] * sc + bi, 0.f);
            float mx  = fmaxf(fmaxf(v00, v01), fmaxf(v10, v11));
            int pc = (xc >> 1) + 4 * hi + p;
            out[(((long)b * Hd + (row + 1)) * Wd + (pc + 1)) * Cout + n] = (_Float16)mx;
        }
    }
}

// ---------------------------------------------------------------------------
// Layer 4: conv3x3 + BN + ReLU, no pool, unpadded output.
// Wide tile: 1 row x 32 cols x 32 out channels per wave (4 accums);
// B fragments are reused across both M-tiles.
// ---------------------------------------------------------------------------
__global__ __launch_bounds__(256)
void conv4_kernel(const _Float16* __restrict__ act,   // [B][Hin+2][Win+2][Cin]
                  const _Float16* __restrict__ wswz,  // [Cout/16][nkt][32][16]
                  const float* __restrict__ bn_g, const float* __restrict__ bn_b,
                  const float* __restrict__ bn_m, const float* __restrict__ bn_v,
                  _Float16* __restrict__ out,         // [B][Hin][Win][Cout]
                  int B, int Hin, int Win, int Cin, int Cout) {
    const int Wp = Win + 2;
    const int lane = threadIdx.x & 31;
    const int lo = lane & 15, hi = lane >> 4;
    const int kb = hi * 8;
    const int wave = (blockIdx.x * blockDim.x + threadIdx.x) >> 5;

    const int ntiles = Cout >> 5;       // 32 channels, full 32-col row per wave
    int t = wave;
    const int nt  = t % ntiles; t /= ntiles;
    const int row = t % Hin;    t /= Hin;
    const int b   = t;
    if (b >= B) return;

    const int n0 = nt * 32 + lo, n1 = n0 + 16;
    const int nkt = (9 * Cin) >> 5;

    const _Float16* wb0 = wswz + ((size_t)(2 * nt) * nkt * 32 + lane) * 16;
    const _Float16* wb1 = wb0 + (size_t)nkt * 512;

    v8f accA0 = {}, accA1 = {}, accB0 = {}, accB1 = {};
    int ktile = 0;
    for (int pos = 0; pos < 9; ++pos) {
        const int ky = pos / 3, kx = pos % 3;
        const _Float16* arowA =
            act + (((size_t)b * (Hin + 2) + (row + ky)) * Wp + (lo + kx)) * Cin;
        const _Float16* arowB = arowA + (size_t)16 * Cin;   // cols 16..31
        for (int ct = 0; ct < Cin; ct += 32, ++ktile) {
            const _Float16* wp0 = wb0 + (size_t)ktile * 512;
            const _Float16* wp1 = wb1 + (size_t)ktile * 512;
            __builtin_prefetch(wp0 + 512, 0, 1);
            v16h bf0 = cat16(*(const v8h*)wp0, *(const v8h*)(wp0 + 8));
            v16h bf1 = cat16(*(const v8h*)wp1, *(const v8h*)(wp1 + 8));

            v16h aA = cat16(*(const v8h*)(arowA + ct + kb),
                            *(const v8h*)(arowA + ct + 16 + kb));
            v16h aB = cat16(*(const v8h*)(arowB + ct + kb),
                            *(const v8h*)(arowB + ct + 16 + kb));
            accA0 = wmma_f16(aA, bf0, accA0);
            accA1 = wmma_f16(aA, bf1, accA1);
            accB0 = wmma_f16(aB, bf0, accB0);
            accB1 = wmma_f16(aB, bf1, accB1);
        }
    }

    #pragma unroll
    for (int g = 0; g < 2; ++g) {
        const int n = g ? n1 : n0;
        const v8f& mA = g ? accA1 : accA0;
        const v8f& mB = g ? accB1 : accB0;
        const float sc = bn_g[n] * rsqrtf(bn_v[n] + 1e-3f);
        const float bi = bn_b[n] - bn_m[n] * sc;
        #pragma unroll
        for (int r = 0; r < 8; ++r) {
            int xA = 8 * hi + r, xB = 16 + xA;
            float vA = fmaxf(mA[r] * sc + bi, 0.f);
            float vB = fmaxf(mB[r] * sc + bi, 0.f);
            out[(((long)b * Hin + row) * Win + xA) * Cout + n] = (_Float16)vA;
            out[(((long)b * Hin + row) * Win + xB) * Cout + n] = (_Float16)vB;
        }
    }
}

// ---------------------------------------------------------------------------
// Global average pool: feat[b,c] = mean(h4[b,:,:,c])
// ---------------------------------------------------------------------------
__global__ void gap_kernel(const _Float16* __restrict__ h, float* __restrict__ feat,
                           int B, int HW, int C) {
    int i = blockIdx.x * blockDim.x + threadIdx.x;
    if (i >= B * C) return;
    int b = i / C, c = i % C;
    const _Float16* p = h + (long)b * HW * C + c;
    float s = 0.f;
    for (int k = 0; k < HW; ++k) s += (float)p[(long)k * C];
    feat[i] = s / (float)HW;
}

// ---------------------------------------------------------------------------
// Dense FC: out[b,j] = (relu?)(sum_k in[b,k]*w[k,j] + bias[j])
// ---------------------------------------------------------------------------
__global__ void fc_kernel(const float* __restrict__ in, const float* __restrict__ w,
                          const float* __restrict__ bias, float* __restrict__ out,
                          int B, int In, int Out, int do_relu) {
    int i = blockIdx.x * blockDim.x + threadIdx.x;
    if (i >= B * Out) return;
    int b = i / Out, j = i % Out;
    float s = bias[j];
    for (int k = 0; k < In; ++k) s += in[b * In + k] * w[k * Out + j];
    out[i] = do_relu ? fmaxf(s, 0.f) : s;
}

// ---------------------------------------------------------------------------
// TPS fiducial control points
// ---------------------------------------------------------------------------
__device__ __forceinline__ void ctrl_point(int k, float* cx, float* cy) {
    int i = k % 10;
    *cx = -1.f + 2.f * (float)i / 9.f;
    *cy = (k < 10) ? -1.f : 1.f;
}

// ---------------------------------------------------------------------------
// Single-block TPS solve: build delta (23x23), Gauss-Jordan invert in LDS,
// then T[b] = inv_delta @ [c_prime[b]; 0]  ->  T: [B, 23, 2]
// ---------------------------------------------------------------------------
__global__ __launch_bounds__(256)
void tps_solve_kernel(const float* __restrict__ cp, float* __restrict__ T, int B) {
    __shared__ float aug[23][47];
    __shared__ float fac[23];
    const int tid = threadIdx.x;

    for (int idx = tid; idx < 23 * 46; idx += 256) {
        int i = idx / 46, j = idx % 46;
        float val = 0.f;
        if (j >= 23) {
            val = (j - 23 == i) ? 1.f : 0.f;
        } else if (i < 20) {
            float xi, yi; ctrl_point(i, &xi, &yi);
            if (j == 0) val = 1.f;
            else if (j == 1) val = xi;
            else if (j == 2) val = yi;
            else {
                int k = j - 3;
                if (k != i) {
                    float xk, yk; ctrl_point(k, &xk, &yk);
                    float dx = xi - xk, dy = yi - yk;
                    float d2 = dx * dx + dy * dy;
                    val = d2 * 0.5f * logf(d2);   // d^2 * ln d
                }
            }
        } else if (i == 20) {
            if (j >= 3) { float x, y; ctrl_point(j - 3, &x, &y); val = x; }
        } else if (i == 21) {
            if (j >= 3) { float x, y; ctrl_point(j - 3, &x, &y); val = y; }
        } else {
            if (j >= 3) val = 1.f;
        }
        aug[i][j] = val;
    }
    __syncthreads();

    for (int p = 0; p < 23; ++p) {
        float piv = aug[p][p];
        __syncthreads();
        float inv = 1.f / piv;
        for (int j = tid; j < 46; j += 256) aug[p][j] *= inv;
        __syncthreads();
        for (int r = tid; r < 23; r += 256) fac[r] = aug[r][p];
        __syncthreads();
        for (int idx = tid; idx < 22 * 46; idx += 256) {
            int r = idx / 46; if (r >= p) r += 1;
            int j = idx % 46;
            aug[r][j] -= fac[r] * aug[p][j];
        }
        __syncthreads();
    }

    for (int idx = tid; idx < B * 23 * 2; idx += 256) {
        int k = idx & 1;
        int rest = idx >> 1;
        int i = rest % 23, b = rest / 23;
        float s = 0.f;
        for (int j = 0; j < 20; ++j)
            s += aug[i][23 + j] * cp[b * 40 + j * 2 + k];
        T[(b * 23 + i) * 2 + k] = s;
    }
}

// ---------------------------------------------------------------------------
// Grid generation (P_hat row on the fly) + bilinear sampling.
// ---------------------------------------------------------------------------
__global__ __launch_bounds__(256)
void tps_sample_kernel(const float* __restrict__ img, const float* __restrict__ T,
                       float* __restrict__ out) {
    __shared__ float Ts[46];
    __shared__ float Cx[20], Cy[20];
    const int b  = blockIdx.y;
    const int py = blockIdx.x;
    const int px = threadIdx.x;

    if (threadIdx.x < 46) Ts[threadIdx.x] = T[b * 46 + threadIdx.x];
    if (threadIdx.x < 20) {
        float x, y; ctrl_point(threadIdx.x, &x, &y);
        Cx[threadIdx.x] = x; Cy[threadIdx.x] = y;
    }
    __syncthreads();

    const float gx = (-256.f + 2.f * px + 1.f) / 256.f;
    const float gy = (-64.f  + 2.f * py + 1.f) / 64.f;

    float xs = Ts[0] + gx * Ts[2] + gy * Ts[4];
    float ys = Ts[1] + gx * Ts[3] + gy * Ts[5];
    #pragma unroll
    for (int k = 0; k < 20; ++k) {
        float dx = gx - Cx[k], dy = gy - Cy[k];
        float rn  = sqrtf(dx * dx + dy * dy);
        float rbf = rn * rn * logf(rn + 1e-6f);
        xs += rbf * Ts[(3 + k) * 2 + 0];
        ys += rbf * Ts[(3 + k) * 2 + 1];
    }

    float fx = (xs + 1.f) * 0.5f * 255.f;
    float fy = (ys + 1.f) * 0.5f * 63.f;
    float x0 = floorf(fx), y0 = floorf(fy);
    float wa = (x0 + 1.f - fx) * (y0 + 1.f - fy);
    float wb = (x0 + 1.f - fx) * (fy - y0);
    float wc = (fx - x0) * (y0 + 1.f - fy);
    float wd = (fx - x0) * (fy - y0);
    int x0i = min(max((int)x0, 0), 255), x1i = min(max((int)x0 + 1, 0), 255);
    int y0i = min(max((int)y0, 0), 63),  y1i = min(max((int)y0 + 1, 0), 63);

    const float* base = img + (long)b * 64 * 256 * 3;
    #pragma unroll
    for (int c = 0; c < 3; ++c) {
        float Ia = base[(y0i * 256 + x0i) * 3 + c];
        float Ib = base[(y1i * 256 + x0i) * 3 + c];
        float Ic = base[(y0i * 256 + x1i) * 3 + c];
        float Id = base[(y1i * 256 + x1i) * 3 + c];
        out[(((long)b * 64 + py) * 256 + px) * 3 + c] = wa * Ia + wb * Ib + wc * Ic + wd * Id;
    }
}

// ---------------------------------------------------------------------------
// Host-side orchestration
// ---------------------------------------------------------------------------
extern "C" void kernel_launch(void* const* d_in, const int* in_sizes, int n_in,
                              void* d_out, int out_size, void* d_ws, size_t ws_size,
                              hipStream_t stream) {
    (void)in_sizes; (void)n_in; (void)out_size; (void)ws_size;

    const float* x    = (const float*)d_in[0];
    const float* w1   = (const float*)d_in[1];
    const float* bn1g = (const float*)d_in[2],  *bn1b = (const float*)d_in[3];
    const float* bn1m = (const float*)d_in[4],  *bn1v = (const float*)d_in[5];
    const float* w2   = (const float*)d_in[6];
    const float* bn2g = (const float*)d_in[7],  *bn2b = (const float*)d_in[8];
    const float* bn2m = (const float*)d_in[9],  *bn2v = (const float*)d_in[10];
    const float* w3   = (const float*)d_in[11];
    const float* bn3g = (const float*)d_in[12], *bn3b = (const float*)d_in[13];
    const float* bn3m = (const float*)d_in[14], *bn3v = (const float*)d_in[15];
    const float* w4   = (const float*)d_in[16];
    const float* bn4g = (const float*)d_in[17], *bn4b = (const float*)d_in[18];
    const float* bn4m = (const float*)d_in[19], *bn4v = (const float*)d_in[20];
    const float* fc1w = (const float*)d_in[21], *fc1b = (const float*)d_in[22];
    const float* fc2w = (const float*)d_in[23], *fc2b = (const float*)d_in[24];
    float* out = (float*)d_out;

    // --- workspace layout -------------------------------------------------
    char* ws = (char*)d_ws;
    size_t off = 0;
    auto alloc = [&](size_t bytes) -> void* {
        void* p = ws + off;
        off = (off + bytes + 255) & ~(size_t)255;
        return p;
    };
    const int B = 64;
    const long n_act0p = (long)B * 66 * 258 * 3;
    const long n_h1p   = (long)B * 34 * 130 * 64;
    const long n_h2p   = (long)B * 18 * 66  * 128;
    const long n_h3p   = (long)B * 10 * 34  * 256;
    const long n_h4    = (long)B * 8  * 32  * 512;
    _Float16* act0p = (_Float16*)alloc(n_act0p * 2);
    _Float16* h1p   = (_Float16*)alloc(n_h1p * 2);
    _Float16* h2p   = (_Float16*)alloc(n_h2p * 2);
    _Float16* h3p   = (_Float16*)alloc(n_h3p * 2);
    _Float16* h4    = (_Float16*)alloc(n_h4 * 2);
    _Float16* ws1 = (_Float16*)alloc((size_t)4  * 1  * 512 * 2);
    _Float16* ws2 = (_Float16*)alloc((size_t)8  * 18 * 512 * 2);
    _Float16* ws3 = (_Float16*)alloc((size_t)16 * 36 * 512 * 2);
    _Float16* ws4 = (_Float16*)alloc((size_t)32 * 72 * 512 * 2);
    float* feat = (float*)alloc((size_t)B * 512 * 4);
    float* f1   = (float*)alloc((size_t)B * 256 * 4);
    float* cp   = (float*)alloc((size_t)B * 40  * 4);
    float* T    = (float*)alloc((size_t)B * 46  * 4);

    // --- zero the haloed activation buffers -------------------------------
    auto zfill = [&](_Float16* p, long n) {
        long n8 = n / 8;
        zero_v8h_kernel<<<(unsigned)((n8 + 255) / 256), 256, 0, stream>>>((v8h*)p, n8);
    };
    zfill(act0p, n_act0p);
    zfill(h1p, n_h1p);
    zfill(h2p, n_h2p);
    zfill(h3p, n_h3p);

    // --- input convert + weight swizzles ----------------------------------
    pad_cvt_kernel<<<(B * 64 * 256 * 3 + 255) / 256, 256, 0, stream>>>(
        x, act0p, B, 64, 256, 3);
    auto swz = [&](const float* w, _Float16* dst, int K, int Cout, int nkt) {
        int total = (Cout >> 4) * nkt * 512;
        wswz_kernel<<<(total + 255) / 256, 256, 0, stream>>>(w, dst, K, Cout, nkt);
    };
    swz(w1, ws1, 27,   64,  1);
    swz(w2, ws2, 576,  128, 18);
    swz(w3, ws3, 1152, 256, 36);
    swz(w4, ws4, 2304, 512, 72);

    // --- localization CNN (WMMA implicit GEMM) ----------------------------
    // conv1: waves = 64*32*16*4 = 131072
    conv1_kernel<<<131072 / 8, 256, 0, stream>>>(
        act0p, ws1, bn1g, bn1b, bn1m, bn1v, h1p);
    // conv2: waves = 64*16*8*(128/32=4) = 32768
    conv_pool_kernel<<<32768 / 8, 256, 0, stream>>>(
        h1p, ws2, bn2g, bn2b, bn2m, bn2v, h2p, B, 32, 128, 64, 128);
    // conv3: waves = 64*8*4*(256/32=8) = 16384
    conv_pool_kernel<<<16384 / 8, 256, 0, stream>>>(
        h2p, ws3, bn3g, bn3b, bn3m, bn3v, h3p, B, 16, 64, 128, 256);
    // conv4: waves = 64*8*(512/32=16) = 8192
    conv4_kernel<<<8192 / 8, 256, 0, stream>>>(
        h3p, ws4, bn4g, bn4b, bn4m, bn4v, h4, B, 8, 32, 256, 512);

    // --- GAP + FC head -----------------------------------------------------
    gap_kernel<<<(B * 512 + 255) / 256, 256, 0, stream>>>(h4, feat, B, 8 * 32, 512);
    fc_kernel<<<(B * 256 + 255) / 256, 256, 0, stream>>>(feat, fc1w, fc1b, f1, B, 512, 256, 1);
    fc_kernel<<<(B * 40  + 255) / 256, 256, 0, stream>>>(f1,   fc2w, fc2b, cp, B, 256, 40, 0);

    // --- TPS solve + grid-gen + bilinear sample ----------------------------
    tps_solve_kernel<<<1, 256, 0, stream>>>(cp, T, B);
    tps_sample_kernel<<<dim3(64, 64), 256, 0, stream>>>(x, T, out);
}
